// BrainEncoder_78761110274172
// MI455X (gfx1250) — matrix operand order repaired and checked by reading, not compiled, not run on hardware
//
#include <hip/hip_runtime.h>

#define NVEC   8192   // B*T
#define NCODE  8192
#define DIM    512
#define Bdim   32
#define Tdim   256

#define ROWS_PER_WG    64
#define CODES_PER_BLK  64
#define LSTR           (DIM + 8)   // padded LDS row stride (ushorts); 1040B rows keep 16B alignment

typedef __attribute__((ext_vector_type(16))) __bf16       v16bf;
typedef __attribute__((ext_vector_type(8)))  float        v8f;
typedef __attribute__((ext_vector_type(4)))  unsigned int u32x4;

union FragU { v16bf bf; u32x4 u[2]; };

static __device__ __forceinline__ unsigned short f2bf(float f) {
  unsigned int u = __float_as_uint(f);
  u += 0x7FFFu + ((u >> 16) & 1u);      // round-to-nearest-even
  return (unsigned short)(u >> 16);
}
static __device__ __forceinline__ float bf2f(unsigned short h) {
  return __uint_as_float(((unsigned int)h) << 16);
}

// ---------------- codebook row norms ----------------
__global__ __launch_bounds__(256) void cnorm_kernel(const float* __restrict__ cb,
                                                    float* __restrict__ cn) {
  const int wid  = threadIdx.x >> 5;
  const int lane = threadIdx.x & 31;
  const int row  = blockIdx.x * 8 + wid;
  if (row >= NCODE) return;
  const float* r = cb + (size_t)row * DIM;
  float s = 0.f;
  #pragma unroll 4
  for (int i = lane; i < DIM; i += 32) { float x = r[i]; s += x * x; }
  #pragma unroll
  for (int off = 16; off >= 1; off >>= 1) s += __shfl_xor(s, off, 32);
  if (lane == 0) cn[row] = s;
}

// ---------------- main WMMA argmin (bf16x3 split-precision GEMM) ----------------
// WG = 128 threads (4 waves) tiled 2x2 over a 64-row x 64-code block.
// Each wave owns a 32x32 output tile: 4 accumulators, 12 WMMAs per 16 ds_load_b128.
__global__ __launch_bounds__(128) void argmin_kernel(const float* __restrict__ z_e,
                                                     const float* __restrict__ cb,
                                                     const float* __restrict__ cn,
                                                     int* __restrict__ idx_out) {
  extern __shared__ unsigned short smem[];
  unsigned short* zhi = smem;                          // 64 x LSTR
  unsigned short* zlo = smem + 64  * LSTR;             // 64 x LSTR
  unsigned short* chi = smem + 128 * LSTR;             // 64 x LSTR
  unsigned short* clo = smem + 192 * LSTR;             // 64 x LSTR
  float*          cnS   = (float*)(smem + 256 * LSTR); // 64 floats
  float*          bestS = cnS + 64;                    // 64 rows x 2 code-halves
  int*            bidxS = (int*)(bestS + 128);         // 64 rows x 2 code-halves

  const int tid = threadIdx.x;
  const int n0  = blockIdx.x * ROWS_PER_WG;
  const int bb  = n0 >> 8;          // batch index (64 | 256 so strip stays in one b)
  const int t0  = n0 & (Tdim - 1);

  // Load + convert the Z strip: for each d, 64 consecutive t (coalesced)
  for (int i = tid; i < ROWS_PER_WG * DIM; i += 128) {
    int d = i >> 6;                 // 0..511
    int r = i & 63;                 // row within strip (t offset)
    float x = z_e[((size_t)bb * DIM + d) * Tdim + (t0 + r)];
    unsigned short h = f2bf(x);
    zhi[r * LSTR + d] = h;
    zlo[r * LSTR + d] = f2bf(x - bf2f(h));
  }
  __syncthreads();

  const int wid   = tid >> 5;
  const int lane  = tid & 31;
  const int laneL = lane & 15;
  const int half  = lane >> 4;
  const int wr    = wid & 1;        // row-tile of this wave   (0,1)
  const int wc    = wid >> 1;       // code-tile of this wave  (0,1)

  float best[2][8];
  int   bidx[2][8];
  #pragma unroll
  for (int s = 0; s < 2; ++s)
    #pragma unroll
    for (int v = 0; v < 8; ++v) { best[s][v] = 3.4e38f; bidx[s][v] = 0; }

  // Per-lane fragment base pointers, per ISA VGPR layouts (wave32):
  //  A (16x32 bf16): lanes 0-15 hold row m, K {k0..k0+7, k0+16..k0+23}; lanes 16-31: {+8, +24}
  //  B (32x16 bf16): lanes 0-15 hold col n, K {k0..k0+7, k0+16..k0+23}; lanes 16-31: {+8, +24}
  const unsigned short* zhiW0 = zhi + (wr * 32 +      laneL) * LSTR;
  const unsigned short* zhiW1 = zhi + (wr * 32 + 16 + laneL) * LSTR;
  const unsigned short* zloW0 = zlo + (wr * 32 +      laneL) * LSTR;
  const unsigned short* zloW1 = zlo + (wr * 32 + 16 + laneL) * LSTR;
  const unsigned short* chiW0 = chi + (wc * 32 +      laneL) * LSTR;
  const unsigned short* chiW1 = chi + (wc * 32 + 16 + laneL) * LSTR;
  const unsigned short* cloW0 = clo + (wc * 32 +      laneL) * LSTR;
  const unsigned short* cloW1 = clo + (wc * 32 + 16 + laneL) * LSTR;
  const int aoff0 = half * 8;
  const int aoff1 = 16 + half * 8;
  const int boff0 = half * 8;
  const int boff1 = 16 + half * 8;

  for (int j0 = 0; j0 < NCODE; j0 += CODES_PER_BLK) {
    // cooperative load of 64 codebook rows, split to bf16 hi/lo
    for (int i = tid; i < CODES_PER_BLK * DIM; i += 128) {
      int r = i >> 9, c = i & 511;
      float x = cb[(size_t)(j0 + r) * DIM + c];
      unsigned short h = f2bf(x);
      chi[r * LSTR + c] = h;
      clo[r * LSTR + c] = f2bf(x - bf2f(h));
    }
    if (tid < CODES_PER_BLK) cnS[tid] = cn[j0 + tid];
    // prefetch next block into L2 (128B lines, 1024 lines per 128KB block)
    {
      int jn = (j0 + CODES_PER_BLK) & (NCODE - 1);
      const float* nb = cb + (size_t)jn * DIM;
      #pragma unroll
      for (int p = 0; p < 8; ++p)
        __builtin_prefetch(nb + (tid + p * 128) * 32, 0, 1);
    }
    __syncthreads();

    v8f acc[2][2];
    #pragma unroll
    for (int s = 0; s < 2; ++s)
      #pragma unroll
      for (int c = 0; c < 2; ++c)
        acc[s][c] = (v8f){0.f, 0.f, 0.f, 0.f, 0.f, 0.f, 0.f, 0.f};

    #pragma unroll
    for (int k0 = 0; k0 < DIM; k0 += 32) {
      FragU ah0, ah1, al0, al1, bh0, bh1, bl0, bl1;
      ah0.u[0] = *(const u32x4*)(zhiW0 + k0 + aoff0);
      ah0.u[1] = *(const u32x4*)(zhiW0 + k0 + aoff1);
      ah1.u[0] = *(const u32x4*)(zhiW1 + k0 + aoff0);
      ah1.u[1] = *(const u32x4*)(zhiW1 + k0 + aoff1);
      al0.u[0] = *(const u32x4*)(zloW0 + k0 + aoff0);
      al0.u[1] = *(const u32x4*)(zloW0 + k0 + aoff1);
      al1.u[0] = *(const u32x4*)(zloW1 + k0 + aoff0);
      al1.u[1] = *(const u32x4*)(zloW1 + k0 + aoff1);
      bh0.u[0] = *(const u32x4*)(chiW0 + k0 + boff0);
      bh0.u[1] = *(const u32x4*)(chiW0 + k0 + boff1);
      bh1.u[0] = *(const u32x4*)(chiW1 + k0 + boff0);
      bh1.u[1] = *(const u32x4*)(chiW1 + k0 + boff1);
      bl0.u[0] = *(const u32x4*)(cloW0 + k0 + boff0);
      bl0.u[1] = *(const u32x4*)(cloW0 + k0 + boff1);
      bl1.u[0] = *(const u32x4*)(cloW1 + k0 + boff0);
      bl1.u[1] = *(const u32x4*)(cloW1 + k0 + boff1);
      // dot ~= hi*hi + hi*lo + lo*hi  (fp32-equivalent precision for argmin)
      acc[0][0] = __builtin_amdgcn_wmma_f32_16x16x32_bf16(false, ah0.bf, false, bh0.bf, (short)0, acc[0][0], false, false);
      acc[0][1] = __builtin_amdgcn_wmma_f32_16x16x32_bf16(false, ah0.bf, false, bh1.bf, (short)0, acc[0][1], false, false);
      acc[1][0] = __builtin_amdgcn_wmma_f32_16x16x32_bf16(false, ah1.bf, false, bh0.bf, (short)0, acc[1][0], false, false);
      acc[1][1] = __builtin_amdgcn_wmma_f32_16x16x32_bf16(false, ah1.bf, false, bh1.bf, (short)0, acc[1][1], false, false);
      acc[0][0] = __builtin_amdgcn_wmma_f32_16x16x32_bf16(false, ah0.bf, false, bl0.bf, (short)0, acc[0][0], false, false);
      acc[0][1] = __builtin_amdgcn_wmma_f32_16x16x32_bf16(false, ah0.bf, false, bl1.bf, (short)0, acc[0][1], false, false);
      acc[1][0] = __builtin_amdgcn_wmma_f32_16x16x32_bf16(false, ah1.bf, false, bl0.bf, (short)0, acc[1][0], false, false);
      acc[1][1] = __builtin_amdgcn_wmma_f32_16x16x32_bf16(false, ah1.bf, false, bl1.bf, (short)0, acc[1][1], false, false);
      acc[0][0] = __builtin_amdgcn_wmma_f32_16x16x32_bf16(false, al0.bf, false, bh0.bf, (short)0, acc[0][0], false, false);
      acc[0][1] = __builtin_amdgcn_wmma_f32_16x16x32_bf16(false, al0.bf, false, bh1.bf, (short)0, acc[0][1], false, false);
      acc[1][0] = __builtin_amdgcn_wmma_f32_16x16x32_bf16(false, al1.bf, false, bh0.bf, (short)0, acc[1][0], false, false);
      acc[1][1] = __builtin_amdgcn_wmma_f32_16x16x32_bf16(false, al1.bf, false, bh1.bf, (short)0, acc[1][1], false, false);
    }

    #pragma unroll
    for (int s = 0; s < 2; ++s)
      #pragma unroll
      for (int c = 0; c < 2; ++c) {
        const int   jj  = j0 + wc * 32 + c * 16 + laneL;
        const float cnv = cnS[wc * 32 + c * 16 + laneL];
        #pragma unroll
        for (int v = 0; v < 8; ++v) {
          float dist = cnv - 2.0f * acc[s][c][v];   // ||z||^2 dropped: constant per row
          if (dist < best[s][v]) { best[s][v] = dist; bidx[s][v] = jj; }
        }
      }
    __syncthreads();
  }

  // lane-reduce min across the 16 column lanes (halves stay separate for xor<16),
  // stash per (row, code-half wave) into LDS
  #pragma unroll
  for (int s = 0; s < 2; ++s)
    #pragma unroll
    for (int v = 0; v < 8; ++v) {
      float b = best[s][v]; int bi = bidx[s][v];
      #pragma unroll
      for (int off = 1; off < 16; off <<= 1) {
        float ob = __shfl_xor(b,  off, 32);
        int   oi = __shfl_xor(bi, off, 32);
        if (ob < b || (ob == b && oi < bi)) { b = ob; bi = oi; }
      }
      if (laneL == 0) {
        int rowLocal = wr * 32 + s * 16 + half * 8 + v;   // D layout: VGPR v = row v / 8+v
        bestS[rowLocal * 2 + wc] = b;
        bidxS[rowLocal * 2 + wc] = bi;
      }
    }
  __syncthreads();

  // combine the two code halves per row and emit indices
  if (tid < ROWS_PER_WG) {
    float b0 = bestS[tid * 2 + 0], b1 = bestS[tid * 2 + 1];
    int   i0 = bidxS[tid * 2 + 0], i1 = bidxS[tid * 2 + 1];
    int bi = (b1 < b0 || (b1 == b0 && i1 < i0)) ? i1 : i0;
    idx_out[n0 + tid] = bi;
  }
}

// ---------------- gather z_q (back to (B,D,T)) + e_latent loss ----------------
__global__ __launch_bounds__(256) void gather_kernel(const float* __restrict__ z_e,
                                                     const float* __restrict__ cb,
                                                     const int* __restrict__ idx,
                                                     float* __restrict__ out,
                                                     float* __restrict__ lossSum) {
  int i  = blockIdx.x * 256 + threadIdx.x;      // index into (B,D,T)
  int t  = i & (Tdim - 1);
  int bd = i >> 8;
  int d  = bd & (DIM - 1);
  int b  = bd >> 9;
  int n  = b * Tdim + t;
  float zq = cb[(size_t)idx[n] * DIM + d];
  float z  = z_e[i];
  out[i] = zq;                                   // straight-through: value == z_q
  float df = zq - z;
  float s = df * df;
  #pragma unroll
  for (int off = 16; off >= 1; off >>= 1) s += __shfl_xor(s, off, 32);
  if ((threadIdx.x & 31) == 0) atomicAdd(lossSum, s);
}

// ---------------- histogram + finalize ----------------
__global__ void init_kernel(float* counts, float* lossSum) {
  int i = blockIdx.x * 256 + threadIdx.x;
  if (i < NCODE) counts[i] = 0.f;
  if (i == 0) lossSum[0] = 0.f;
}

__global__ void hist_kernel(const int* __restrict__ idx, float* __restrict__ counts) {
  int n = blockIdx.x * 256 + threadIdx.x;
  if (n < NVEC) atomicAdd(&counts[idx[n]], 1.0f);
}

__global__ __launch_bounds__(256) void finalize_kernel(const float* __restrict__ counts,
                                                       const float* __restrict__ lossSum,
                                                       float* __restrict__ out) {
  __shared__ float red[8];
  float s = 0.f;
  for (int k = threadIdx.x; k < NCODE; k += 256) {
    float p = counts[k] * (1.0f / (float)NVEC);
    s += p * logf(p + 1e-10f);
  }
  #pragma unroll
  for (int off = 16; off >= 1; off >>= 1) s += __shfl_xor(s, off, 32);
  if ((threadIdx.x & 31) == 0) red[threadIdx.x >> 5] = s;
  __syncthreads();
  if (threadIdx.x == 0) {
    float tot = 0.f;
    #pragma unroll
    for (int w = 0; w < 8; ++w) tot += red[w];
    const size_t NE = (size_t)Bdim * DIM * Tdim;
    out[NE]     = 0.25f * (lossSum[0] / (float)((size_t)NVEC * DIM)); // ALPHA*BETA*mean
    out[NE + 1] = expf(-tot);                                         // perplexity
  }
}

extern "C" void kernel_launch(void* const* d_in, const int* in_sizes, int n_in,
                              void* d_out, int out_size, void* d_ws, size_t ws_size,
                              hipStream_t stream) {
  (void)in_sizes; (void)n_in; (void)out_size; (void)ws_size;
  const float* z_e = (const float*)d_in[0];
  const float* cb  = (const float*)d_in[1];
  float* out = (float*)d_out;

  float* cn      = (float*)d_ws;                 // [0,    8192)
  int*   idx     = (int*)((float*)d_ws + 8192);  // [8192, 16384)
  float* counts  = (float*)d_ws + 16384;         // [16384,24576)
  float* lossSum = (float*)d_ws + 24576;         // [24576]

  init_kernel<<<(NCODE + 255) / 256, 256, 0, stream>>>(counts, lossSum);
  cnorm_kernel<<<NCODE / 8, 256, 0, stream>>>(cb, cn);

  size_t smem = (size_t)256 * LSTR * sizeof(unsigned short)   // z + codebook hi/lo
              + 64 * sizeof(float)                            // cnorm block
              + 128 * sizeof(float) + 128 * sizeof(int);      // cross-wave argmin combine
  argmin_kernel<<<NVEC / ROWS_PER_WG, 128, smem, stream>>>(z_e, cb, cn, idx);

  gather_kernel<<<(Bdim * DIM * Tdim) / 256, 256, 0, stream>>>(z_e, cb, idx, out, lossSum);
  hist_kernel<<<NVEC / 256, 256, 0, stream>>>(idx, counts);
  finalize_kernel<<<1, 256, 0, stream>>>(counts, lossSum, out);
}
